// MultiHeadAttention_78434692760054
// MI455X (gfx1250) — matrix operand order repaired and checked by reading, not compiled, not run on hardware
//
#include <hip/hip_runtime.h>
#include <hip/hip_bf16.h>
#include <stddef.h>

// Problem constants (match reference)
constexpr int CB = 2;      // batch
constexpr int CS = 2048;   // sequence
constexpr int CD = 1024;   // model dim
constexpr int CH = 16;     // heads
constexpr int CDH = 64;    // head dim

typedef __attribute__((ext_vector_type(16))) __bf16 v16bf;
typedef __attribute__((ext_vector_type(8)))  __bf16 v8bf;
typedef __attribute__((ext_vector_type(4)))  __bf16 v4bf;
typedef __attribute__((ext_vector_type(8)))  float  v8f;

__device__ __forceinline__ v8f wmma_bf16(v16bf a, v16bf b, v8f c) {
  return __builtin_amdgcn_wmma_f32_16x16x32_bf16(false, a, false, b, (short)0, c,
                                                 false, false);
}

// Two 16-byte loads (ds_load_b128 / global_load_b128) -> one 16-elem fragment.
__device__ __forceinline__ v16bf ld_frag2(const __bf16* p0, const __bf16* p1) {
  v8bf lo = *(const v8bf*)p0;
  v8bf hi = *(const v8bf*)p1;
  return __builtin_shufflevector(lo, hi, 0, 1, 2, 3, 4, 5, 6, 7, 8, 9, 10, 11,
                                 12, 13, 14, 15);
}

__device__ __forceinline__ v4bf cvt4(float4 f) {
  v4bf o;
  o[0] = (__bf16)f.x; o[1] = (__bf16)f.y; o[2] = (__bf16)f.z; o[3] = (__bf16)f.w;
  return o;
}

// ---------------------------------------------------------------------------
// Kernel 1: fused per-head Q/K/V projection (bf16 WMMA, f32 accum).
// grid = 3*B*H*(S/128); block = 256 (8 waves); 128x64 tile per block.
// A staged row-major (lane frag = two contiguous 8-elem runs); B staged
// transposed Bst[n][k] so B-frag reads are contiguous.
// ---------------------------------------------------------------------------
__global__ __launch_bounds__(256) void qkv_proj_kernel(
    const float* __restrict__ x,
    const float* __restrict__ wq, const float* __restrict__ wk,
    const float* __restrict__ wv,
    const float* __restrict__ bq, const float* __restrict__ bk,
    const float* __restrict__ bv,
    __bf16* __restrict__ qo, __bf16* __restrict__ ko, __bf16* __restrict__ vo) {
  __shared__ __attribute__((aligned(16))) __bf16 As[128][40];   // rows x K
  __shared__ __attribute__((aligned(16))) __bf16 Bst[64][40];   // n x K (transposed)

  const int tid = threadIdx.x;
  int t = blockIdx.x;
  const int mt    = t & 15; t >>= 4;
  const int h     = t & 15; t >>= 4;
  const int b     = t & 1;  t >>= 1;
  const int which = t;  // 0=q 1=k 2=v

  const float* w    = (which == 0) ? wq : (which == 1) ? wk : wv;
  const float* bias = (which == 0) ? bq : (which == 1) ? bk : bv;
  __bf16*      outp = (which == 0) ? qo : (which == 1) ? ko : vo;
  w    += (size_t)h * CD * CDH;
  bias += h * CDH;

  const int wid = tid >> 5, lane = tid & 31, lrow = lane & 15, khalf = lane >> 4;

  v8f acc[4];
#pragma unroll
  for (int nt = 0; nt < 4; ++nt)
#pragma unroll
    for (int r = 0; r < 8; ++r) acc[nt][r] = 0.f;

  const float* xb = x + ((size_t)b * CS + (size_t)mt * 128) * CD;

  for (int k0 = 0; k0 < CD; k0 += 32) {
    __syncthreads();
    // x tile 128x32: float4 reads -> v4bf LDS stores
#pragma unroll
    for (int i = 0; i < 4; ++i) {
      int idx4 = tid + i * 256;
      int r = idx4 >> 3, c4 = (idx4 & 7) * 4;
      *(v4bf*)&As[r][c4] = cvt4(*(const float4*)&xb[(size_t)r * CD + k0 + c4]);
    }
    // w tile 32x64 -> transposed Bst[n][kk]
#pragma unroll
    for (int i = 0; i < 2; ++i) {
      int idx4 = tid + i * 256;
      int kk = idx4 >> 4, n4 = (idx4 & 15) * 4;
      float4 f = *(const float4*)&w[(size_t)(k0 + kk) * CDH + n4];
      Bst[n4 + 0][kk] = (__bf16)f.x;
      Bst[n4 + 1][kk] = (__bf16)f.y;
      Bst[n4 + 2][kk] = (__bf16)f.z;
      Bst[n4 + 3][kk] = (__bf16)f.w;
    }
    __syncthreads();

    const int arow = wid * 16 + lrow;
    v16bf a = ld_frag2(&As[arow][8 * khalf], &As[arow][16 + 8 * khalf]);
#pragma unroll
    for (int nt = 0; nt < 4; ++nt) {
      v16bf bfr = ld_frag2(&Bst[nt * 16 + lrow][8 * khalf],
                           &Bst[nt * 16 + lrow][16 + 8 * khalf]);
      acc[nt] = wmma_bf16(a, bfr, acc[nt]);
    }
  }

  // C layout: (VGPR r, lane l) -> M = r + 8*(l>=16), N = l%16
  __bf16* ob = outp + ((size_t)(b * CH + h) * CS + (size_t)mt * 128) * CDH;
#pragma unroll
  for (int nt = 0; nt < 4; ++nt) {
    int n = nt * 16 + lrow;
    float bia = bias[n];
#pragma unroll
    for (int r = 0; r < 8; ++r) {
      int row = wid * 16 + r + 8 * khalf;
      ob[(size_t)row * CDH + n] = (__bf16)(acc[nt][r] + bia);
    }
  }
}

// ---------------------------------------------------------------------------
// Kernel 2: flash attention. grid = B*H*(S/128); block = 256 (8 waves).
// Q frags pinned in VGPRs; K row-major, V transposed in LDS so both B-frag
// reads are contiguous b128 pairs; online softmax; P via per-wave LDS.
// ---------------------------------------------------------------------------
__global__ __launch_bounds__(256) void attn_kernel(
    const __bf16* __restrict__ q, const __bf16* __restrict__ k,
    const __bf16* __restrict__ v, const int* __restrict__ pad_mask,
    __bf16* __restrict__ concat) {
  __shared__ __attribute__((aligned(16))) __bf16 Ks[64][72];      // key x dh
  __shared__ __attribute__((aligned(16))) __bf16 Vt[64][72];      // dh x key
  __shared__ __attribute__((aligned(16))) __bf16 Ps[8][16][72];   // per-wave P

  int t = blockIdx.x;
  const int qt = t & 15; t >>= 4;
  const int h  = t & 15; t >>= 4;
  const int b  = t;
  const int tid = threadIdx.x;
  const int wid = tid >> 5, lane = tid & 31, lrow = lane & 15, khalf = lane >> 4;

  const size_t headoff = (size_t)(b * CH + h) * CS * CDH;
  const int qrow = qt * 128 + wid * 16 + lrow;

  // Q fragments: two global_load_b128 per 16x32 half
  v16bf qf[2];
  const __bf16* qp = q + headoff + (size_t)qrow * CDH;
#pragma unroll
  for (int half = 0; half < 2; ++half)
    qf[half] = ld_frag2(qp + half * 32 + 8 * khalf,
                        qp + half * 32 + 16 + 8 * khalf);

  float rmax[8], rsum[8];
  v8f oacc[4];
#pragma unroll
  for (int r = 0; r < 8; ++r) { rmax[r] = -1e30f; rsum[r] = 0.f; }
#pragma unroll
  for (int nt = 0; nt < 4; ++nt)
#pragma unroll
    for (int r = 0; r < 8; ++r) oacc[nt][r] = 0.f;

  const float scale = 0.125f;  // 1/sqrt(64)

  for (int kb = 0; kb < CS / 64; ++kb) {
    __syncthreads();
#pragma unroll
    for (int i = 0; i < 2; ++i) {  // 64x64 K and V tiles, v8bf granularity
      int idx8 = tid + i * 256;
      int r = idx8 >> 3, c8 = (idx8 & 7) * 8;
      *(v8bf*)&Ks[r][c8] =
          *(const v8bf*)&k[headoff + (size_t)(kb * 64 + r) * CDH + c8];
      v8bf vv = *(const v8bf*)&v[headoff + (size_t)(kb * 64 + r) * CDH + c8];
#pragma unroll
      for (int j = 0; j < 8; ++j) Vt[c8 + j][r] = vv[j];
    }
    // prefetch next K/V block into cache while we compute
    if (kb + 1 < CS / 64) {
      const __bf16* nk = k + headoff + (size_t)(kb + 1) * 64 * CDH + tid * 16;
      const __bf16* nv = v + headoff + (size_t)(kb + 1) * 64 * CDH + tid * 16;
      __builtin_prefetch(nk, 0, 0);
      __builtin_prefetch(nv, 0, 0);
    }
    __syncthreads();

    // scores = Q * K^T : four 16x16 tiles over 64 keys
    v8f sc[4];
#pragma unroll
    for (int nt = 0; nt < 4; ++nt) {
#pragma unroll
      for (int r = 0; r < 8; ++r) sc[nt][r] = 0.f;
#pragma unroll
      for (int half = 0; half < 2; ++half) {
        v16bf bfr = ld_frag2(&Ks[nt * 16 + lrow][half * 32 + 8 * khalf],
                             &Ks[nt * 16 + lrow][half * 32 + 16 + 8 * khalf]);
        sc[nt] = wmma_bf16(qf[half], bfr, sc[nt]);
      }
    }

    // scale + pad-key mask (column-wise: key = kb*64 + nt*16 + lane%16)
#pragma unroll
    for (int nt = 0; nt < 4; ++nt) {
      int key = kb * 64 + nt * 16 + lrow;
      bool valid = pad_mask[b * CS + key] != 0;
#pragma unroll
      for (int r = 0; r < 8; ++r)
        sc[nt][r] = valid ? sc[nt][r] * scale : -1e9f;
    }

    // online softmax: row-wise max across the 16 lanes holding each row
#pragma unroll
    for (int r = 0; r < 8; ++r) {
      float m = fmaxf(fmaxf(sc[0][r], sc[1][r]), fmaxf(sc[2][r], sc[3][r]));
#pragma unroll
      for (int off = 1; off < 16; off <<= 1)
        m = fmaxf(m, __shfl_xor(m, off, 32));
      float nm   = fmaxf(rmax[r], m);
      float corr = __expf(rmax[r] - nm);
      rmax[r] = nm;
      rsum[r] *= corr;
#pragma unroll
      for (int nt = 0; nt < 4; ++nt) oacc[nt][r] *= corr;
    }

    // exponentiate; write P to per-wave LDS (C-layout -> row-major [m][key])
#pragma unroll
    for (int nt = 0; nt < 4; ++nt)
#pragma unroll
      for (int r = 0; r < 8; ++r) {
        float p = __expf(sc[nt][r] - rmax[r]);
        sc[nt][r] = p;
        Ps[wid][r + 8 * khalf][nt * 16 + lrow] = (__bf16)p;
      }

    // row sums
#pragma unroll
    for (int r = 0; r < 8; ++r) {
      float s = sc[0][r] + sc[1][r] + sc[2][r] + sc[3][r];
#pragma unroll
      for (int off = 1; off < 16; off <<= 1) s += __shfl_xor(s, off, 32);
      rsum[r] += s;
    }

    asm volatile("s_wait_dscnt 0" ::: "memory");  // P store->load turnaround

    // O += P(16x64) * V(64x64)
#pragma unroll
    for (int half = 0; half < 2; ++half) {
      v16bf pa = ld_frag2(&Ps[wid][lrow][half * 32 + 8 * khalf],
                          &Ps[wid][lrow][half * 32 + 16 + 8 * khalf]);
#pragma unroll
      for (int nt = 0; nt < 4; ++nt) {
        v16bf bv = ld_frag2(&Vt[nt * 16 + lrow][half * 32 + 8 * khalf],
                            &Vt[nt * 16 + lrow][half * 32 + 16 + 8 * khalf]);
        oacc[nt] = wmma_bf16(pa, bv, oacc[nt]);
      }
    }
  }

  // normalize and store heads directly in concat layout [b][s][h*DH+dh]
#pragma unroll
  for (int nt = 0; nt < 4; ++nt)
#pragma unroll
    for (int r = 0; r < 8; ++r) {
      int row = qt * 128 + wid * 16 + r + 8 * khalf;
      int col = h * CDH + nt * 16 + lrow;
      concat[((size_t)b * CS + row) * CD + col] = (__bf16)(oacc[nt][r] / rsum[r]);
    }
}

// ---------------------------------------------------------------------------
// Kernel 3: output projection out = concat @ wo^T + bo.
// wo is [n][k] row-major == exactly the transposed-B LDS layout we want.
// ---------------------------------------------------------------------------
__global__ __launch_bounds__(256) void out_proj_kernel(
    const __bf16* __restrict__ concat, const float* __restrict__ wo,
    const float* __restrict__ bo, float* __restrict__ out) {
  __shared__ __attribute__((aligned(16))) __bf16 As[128][40];
  __shared__ __attribute__((aligned(16))) __bf16 Bst[64][40];  // n x K

  const int tid = threadIdx.x;
  const int nt0 = blockIdx.x & 15;   // N tile of 64 (D/64 = 16)
  const int mt  = blockIdx.x >> 4;   // 0..31 over B*S/128
  const int wid = tid >> 5, lane = tid & 31, lrow = lane & 15, khalf = lane >> 4;

  v8f acc[4];
#pragma unroll
  for (int nt = 0; nt < 4; ++nt)
#pragma unroll
    for (int r = 0; r < 8; ++r) acc[nt][r] = 0.f;

  for (int k0 = 0; k0 < CD; k0 += 32) {
    __syncthreads();
#pragma unroll
    for (int i = 0; i < 2; ++i) {  // 128x32 bf16 A tile, v8bf copies
      int idx8 = tid + i * 256;
      int r = idx8 >> 2, c8 = (idx8 & 3) * 8;
      *(v8bf*)&As[r][c8] =
          *(const v8bf*)&concat[((size_t)mt * 128 + r) * CD + k0 + c8];
    }
#pragma unroll
    for (int i = 0; i < 2; ++i) {  // 64x32 B tile from wo rows (contiguous)
      int idx4 = tid + i * 256;
      int n = idx4 >> 3, k4 = (idx4 & 7) * 4;
      *(v4bf*)&Bst[n][k4] =
          cvt4(*(const float4*)&wo[(size_t)(nt0 * 64 + n) * CD + k0 + k4]);
    }
    __syncthreads();

    const int arow = wid * 16 + lrow;
    v16bf a = ld_frag2(&As[arow][8 * khalf], &As[arow][16 + 8 * khalf]);
#pragma unroll
    for (int nt = 0; nt < 4; ++nt) {
      v16bf bfr = ld_frag2(&Bst[nt * 16 + lrow][8 * khalf],
                           &Bst[nt * 16 + lrow][16 + 8 * khalf]);
      acc[nt] = wmma_bf16(a, bfr, acc[nt]);
    }
  }

#pragma unroll
  for (int nt = 0; nt < 4; ++nt) {
    int gn = nt0 * 64 + nt * 16 + lrow;
    float bia = bo[gn];
#pragma unroll
    for (int r = 0; r < 8; ++r) {
      int row = mt * 128 + wid * 16 + r + 8 * khalf;
      out[(size_t)row * CD + gn] = acc[nt][r] + bia;
    }
  }
}

// ---------------------------------------------------------------------------
extern "C" void kernel_launch(void* const* d_in, const int* in_sizes, int n_in,
                              void* d_out, int out_size, void* d_ws,
                              size_t ws_size, hipStream_t stream) {
  const float* x        = (const float*)d_in[0];
  const int*   pad_mask = (const int*)d_in[1];
  const float* wq       = (const float*)d_in[2];
  const float* wk       = (const float*)d_in[3];
  const float* wv       = (const float*)d_in[4];
  const float* bq       = (const float*)d_in[5];
  const float* bk       = (const float*)d_in[6];
  const float* bv       = (const float*)d_in[7];
  const float* wo       = (const float*)d_in[8];
  const float* bo       = (const float*)d_in[9];
  float* out = (float*)d_out;

  // Workspace: q,k,v [B,H,S,DH] bf16 + concat [B,S,D] bf16 = 32 MB
  const size_t per = (size_t)CB * CH * CS * CDH;
  __bf16* qb = (__bf16*)d_ws;
  __bf16* kb = qb + per;
  __bf16* vb = kb + per;
  __bf16* cb = vb + per;

  const int qkv_blocks  = 3 * CB * CH * (CS / 128);     // 1536
  const int attn_blocks = CB * CH * (CS / 128);         // 512
  const int proj_blocks = (CB * CS / 128) * (CD / 64);  // 512

  qkv_proj_kernel<<<qkv_blocks, 256, 0, stream>>>(x, wq, wk, wv, bq, bk, bv,
                                                  qb, kb, vb);
  attn_kernel<<<attn_blocks, 256, 0, stream>>>(qb, kb, vb, pad_mask, cb);
  out_proj_kernel<<<proj_blocks, 256, 0, stream>>>(cb, wo, bo, out);
}
